// MultiHeadAttention_14980845929013
// MI455X (gfx1250) — matrix operand order repaired and checked by reading, not compiled, not run on hardware
//
#include <hip/hip_runtime.h>

// ---------------------------------------------------------------------------
// MI455X (gfx1250) fused multi-head attention: bf16 WMMA flash attention.
// V is produced pre-transposed [B,H,Dk,S] so attention needs no LDS transpose
// and no barriers; Q tiles staged via CDNA5 async global->LDS copies.
// ---------------------------------------------------------------------------

typedef __bf16  bf16;
typedef __bf16  bf16x8  __attribute__((ext_vector_type(8)));
typedef __bf16  bf16x16 __attribute__((ext_vector_type(16)));
typedef float   v8f     __attribute__((ext_vector_type(8)));

#define D_MODEL 768
#define NHEADS  12
#define DK      64
#define BATCH   4
#define SEQ     2048
#define TOK     (BATCH * SEQ)         // 8192
#define LOG2E   1.4426950408889634f

// xor-shuffle within the 32-lane wave via ds_swizzle (bitmask mode:
// and=0x1f, or=0, xor=m). Masks 1/2/4/8 stay within each 16-lane half.
#define XORF(v, m) __int_as_float(__builtin_amdgcn_ds_swizzle(__float_as_int(v), (0x1f | ((m) << 10))))

__device__ __forceinline__ float red_max16(float v) {
    v = fmaxf(v, XORF(v, 1)); v = fmaxf(v, XORF(v, 2));
    v = fmaxf(v, XORF(v, 4)); v = fmaxf(v, XORF(v, 8));
    return v;
}
__device__ __forceinline__ float red_sum16(float v) {
    v += XORF(v, 1); v += XORF(v, 2); v += XORF(v, 4); v += XORF(v, 8);
    return v;
}

__device__ __forceinline__ v8f vzero8() {
    v8f z = {0.f, 0.f, 0.f, 0.f, 0.f, 0.f, 0.f, 0.f};
    return z;
}

__device__ __forceinline__ v8f wmma_bf16(bf16x16 a, bf16x16 b, v8f c) {
    return __builtin_amdgcn_wmma_f32_16x16x32_bf16(
        /*neg_a=*/false, a, /*neg_b=*/false, b,
        /*c_mod=*/(short)0, c, /*reuse_a=*/false, /*reuse_b=*/false);
}

// A fragment (16x32 bf16, M x K). ISA 7.12.2: lane L holds row m = m0+(L&15);
// elements 0..7 -> k0 + (L>>4)*8 .. +7 ; elements 8..15 -> +16 more.
__device__ __forceinline__ bf16x16 load_a_frag(const bf16* __restrict__ base,
                                               int lda, int m0, int k0, int lane) {
    int m = m0 + (lane & 15);
    int k = k0 + ((lane >> 4) << 3);
    const bf16* p = base + (size_t)m * lda + k;
    bf16x8 lo = *(const bf16x8*)(p);
    bf16x8 hi = *(const bf16x8*)(p + 16);
    return __builtin_shufflevector(lo, hi, 0,1,2,3,4,5,6,7,8,9,10,11,12,13,14,15);
}

// B fragment (32x16 bf16, K x N) where B[k][n] = src[n][k] (src row-major NxK).
// Lane L holds column n = n0+(L&15); elements 0..7 -> k0 + (L>>4)*8 .. +7 ; elements 8..15 -> +16 more.
__device__ __forceinline__ bf16x16 load_b_frag(const bf16* __restrict__ base,
                                               int ldb, int n0, int k0, int lane) {
    int n = n0 + (lane & 15);
    int k = k0 + ((lane >> 4) << 3);
    const bf16* p = base + (size_t)n * ldb + k;
    bf16x8 lo = *(const bf16x8*)(p);
    bf16x8 hi = *(const bf16x8*)(p + 16);
    return __builtin_shufflevector(lo, hi, 0,1,2,3,4,5,6,7,8,9,10,11,12,13,14,15);
}

// ---------------------------------------------------------------------------
// fp32 -> bf16 conversion
// ---------------------------------------------------------------------------
__global__ __launch_bounds__(256) void cvt_f32_bf16(const float* __restrict__ src,
                                                    bf16* __restrict__ dst, int n) {
    int i = blockIdx.x * 256 + threadIdx.x;
    if (i < n) dst[i] = (bf16)src[i];
}

// ---------------------------------------------------------------------------
// GEMM: C[M x N] = A[M x K] @ W[N x K]^T + bias, bf16 in, fp32 accumulate.
// Block = 256 threads (8 waves), tile 128(M) x 64(N); wave = 32x32 (2x2 WMMA).
// mode 0: bf16 scatter to [B,H,S,Dk]   (Q, K)
// mode 1: fp32 row-major [M x N]       (final output)
// mode 2: bf16 scatter to [B,H,Dk,S]   (V, pre-transposed for attention)
// ---------------------------------------------------------------------------
__global__ __launch_bounds__(256) void gemm_bf16_kernel(
    const bf16* __restrict__ A, const bf16* __restrict__ W,
    const float* __restrict__ bias, void* __restrict__ out,
    int M, int N, int K, int mode, float scale) {

    const int lane = threadIdx.x & 31;
    const int wave = threadIdx.x >> 5;
    const int wm = wave & 3, wn = wave >> 2;
    const int m0 = blockIdx.x * 128 + wm * 32;
    const int n0 = blockIdx.y * 64 + wn * 32;

    v8f acc[2][2];
    acc[0][0] = vzero8(); acc[0][1] = vzero8();
    acc[1][0] = vzero8(); acc[1][1] = vzero8();

    for (int k0 = 0; k0 < K; k0 += 32) {
        bf16x16 a0 = load_a_frag(A, K, m0,      k0, lane);
        bf16x16 a1 = load_a_frag(A, K, m0 + 16, k0, lane);
        bf16x16 b0 = load_b_frag(W, K, n0,      k0, lane);
        bf16x16 b1 = load_b_frag(W, K, n0 + 16, k0, lane);
        acc[0][0] = wmma_bf16(a0, b0, acc[0][0]);
        acc[0][1] = wmma_bf16(a0, b1, acc[0][1]);
        acc[1][0] = wmma_bf16(a1, b0, acc[1][0]);
        acc[1][1] = wmma_bf16(a1, b1, acc[1][1]);
    }

    const int nlo  = lane & 15;
    const int half = lane >> 4;
#pragma unroll
    for (int ms = 0; ms < 2; ++ms) {
#pragma unroll
        for (int ns = 0; ns < 2; ++ns) {
            const int n  = n0 + ns * 16 + nlo;
            const float bv = bias[n];
#pragma unroll
            for (int r = 0; r < 8; ++r) {
                const int m = m0 + ms * 16 + r + 8 * half;
                float v = (acc[ms][ns][r] + bv) * scale;
                if (mode == 1) {
                    ((float*)out)[(size_t)m * N + n] = v;
                } else {
                    const int bi = m >> 11, s = m & 2047;
                    const int h = n >> 6, dk = n & 63;
                    size_t oi;
                    if (mode == 0)       // [B, H, S, Dk]
                        oi = (((size_t)(bi * NHEADS + h) * SEQ) + s) * DK + dk;
                    else                 // [B, H, Dk, S]
                        oi = (((size_t)(bi * NHEADS + h) * DK) + dk) * SEQ + s;
                    ((bf16*)out)[oi] = (bf16)v;
                }
            }
        }
    }
}

// ---------------------------------------------------------------------------
// Flash attention: Q,K bf16 [B,H,S,Dk]; V bf16 [B,H,Dk,S]; O bf16 [B,S,H*Dk].
// Block = 256 threads = 8 fully independent waves (no barriers); block handles
// (b, h, 128 query rows); each wave owns 16 query rows. 64-key KV tiles,
// online softmax. Q staged through LDS with CDNA5 async global->LDS copies.
// ---------------------------------------------------------------------------
#define KT 64

__global__ __launch_bounds__(256) void flash_attn_kernel(
    const bf16* __restrict__ Q, const bf16* __restrict__ Km,
    const bf16* __restrict__ Vm, bf16* __restrict__ O) {

    __shared__ __align__(32) bf16 Qlds[8][16][DK];   // per-wave Q tile (16KB)
    __shared__ __align__(32) bf16 Pbuf[8][16][KT];   // per-wave P staging (16KB)

    const int lane = threadIdx.x & 31;
    const int wave = threadIdx.x >> 5;
    const int half = lane >> 4;
    const int h = blockIdx.y, b = blockIdx.z;

    const bf16* Qh = Q  + (size_t)(b * NHEADS + h) * SEQ * DK;
    const bf16* Kh = Km + (size_t)(b * NHEADS + h) * SEQ * DK;
    const bf16* Vh = Vm + (size_t)(b * NHEADS + h) * DK * SEQ;   // [Dk][S]

    const int q0 = blockIdx.x * 128 + wave * 16;

    // ---- async-stage this wave's Q tile (16 x 64 bf16 = 2KB) into LDS ----
    // 128 16-byte chunks; 32 lanes x 4 issues. ASYNCcnt-tracked.
    {
        const unsigned lbase = (unsigned)(size_t)&Qlds[wave][0][0];
#pragma unroll
        for (int i = 0; i < 4; ++i) {
            const int chunk = i * 32 + lane;           // 0..127
            const int row = chunk >> 3;                // 0..15
            const int col = (chunk & 7) << 3;          // 0,8,..,56
            const unsigned long long ga =
                (unsigned long long)(size_t)(Qh + (size_t)(q0 + row) * DK + col);
            const unsigned la = lbase + (unsigned)(chunk << 4);
            asm volatile("global_load_async_to_lds_b128 %0, %1, off"
                         :: "v"(la), "v"(ga) : "memory");
        }
        asm volatile("s_wait_asynccnt 0x0" ::: "memory");
    }

    // Q fragments for this wave's 16 rows (d = 0..31 and 32..63), A-layout.
    const bf16* Ql = &Qlds[wave][0][0];
    bf16x16 qf0 = load_a_frag(Ql, DK, 0, 0,  lane);
    bf16x16 qf1 = load_a_frag(Ql, DK, 0, 32, lane);

    v8f oacc[4];
    oacc[0] = vzero8(); oacc[1] = vzero8(); oacc[2] = vzero8(); oacc[3] = vzero8();
    float mrun[8], lrun[8];
#pragma unroll
    for (int r = 0; r < 8; ++r) { mrun[r] = -1e30f; lrun[r] = 0.f; }

    for (int kv0 = 0; kv0 < SEQ; kv0 += KT) {
        // ---- prefetch next KV tile into cache (global_prefetch_b8) ----
        if (kv0 + KT < SEQ) {
            if (wave == 0) {
                // next K tile: 64 rows x 128B
                __builtin_prefetch(Kh + (size_t)(kv0 + KT + lane * 2) * DK, 0, 1);
            } else if (wave == 1) {
                // next V tile: 64 rows x 128B (transposed layout)
                __builtin_prefetch(Vh + (size_t)(lane * 2) * SEQ + kv0 + KT, 0, 1);
                __builtin_prefetch(Vh + (size_t)(lane * 2 + 1) * SEQ + kv0 + KT, 0, 1);
            }
        }

        // ---- scores: S(16 x 64) = Q(16 x 64) @ K^T, 4 key sub-tiles ----
        v8f sacc[4];
#pragma unroll
        for (int nt = 0; nt < 4; ++nt) {
            sacc[nt] = vzero8();
            bf16x16 kb0 = load_b_frag(Kh, DK, kv0 + nt * 16, 0,  lane);
            bf16x16 kb1 = load_b_frag(Kh, DK, kv0 + nt * 16, 32, lane);
            sacc[nt] = wmma_bf16(qf0, kb0, sacc[nt]);
            sacc[nt] = wmma_bf16(qf1, kb1, sacc[nt]);
        }

        // ---- online softmax update (rows live across 16-lane halves) ----
        float alpha[8];
#pragma unroll
        for (int r = 0; r < 8; ++r) {
            float tm = fmaxf(fmaxf(sacc[0][r], sacc[1][r]),
                             fmaxf(sacc[2][r], sacc[3][r]));
            tm = red_max16(tm);
            const float mnew = fmaxf(mrun[r], tm);
            alpha[r] = __builtin_exp2f((mrun[r] - mnew) * LOG2E);
            mrun[r] = mnew;
        }

        float rsum[8];
#pragma unroll
        for (int r = 0; r < 8; ++r) rsum[r] = 0.f;
#pragma unroll
        for (int nt = 0; nt < 4; ++nt) {
#pragma unroll
            for (int r = 0; r < 8; ++r) {
                const float p = __builtin_exp2f((sacc[nt][r] - mrun[r]) * LOG2E);
                rsum[r] += p;
                Pbuf[wave][r + 8 * half][nt * 16 + (lane & 15)] = (bf16)p;
            }
        }
#pragma unroll
        for (int r = 0; r < 8; ++r) {
            lrun[r] = lrun[r] * alpha[r] + red_sum16(rsum[r]);
#pragma unroll
            for (int dt = 0; dt < 4; ++dt) oacc[dt][r] *= alpha[r];
        }

        // ---- O += P(16 x 64) @ V(64 x 64); V read directly (pre-transposed) ----
        const bf16* Pl = &Pbuf[wave][0][0];
#pragma unroll
        for (int kc = 0; kc < 2; ++kc) {
            bf16x16 pa = load_a_frag(Pl, KT, 0, kc * 32, lane);
#pragma unroll
            for (int dt = 0; dt < 4; ++dt) {
                bf16x16 vb = load_b_frag(Vh, SEQ, dt * 16, kv0 + kc * 32, lane);
                oacc[dt] = wmma_bf16(pa, vb, oacc[dt]);
            }
        }
    }

    // ---- normalize and write O as [B, S, H*Dk] bf16 ----
#pragma unroll
    for (int dt = 0; dt < 4; ++dt) {
#pragma unroll
        for (int r = 0; r < 8; ++r) {
            const int s   = q0 + r + 8 * half;
            const int col = h * DK + dt * 16 + (lane & 15);
            const float v = oacc[dt][r] / lrun[r];
            O[(size_t)(b * SEQ + s) * D_MODEL + col] = (bf16)v;
        }
    }
}

// ---------------------------------------------------------------------------
// Host-side orchestration
// ---------------------------------------------------------------------------
extern "C" void kernel_launch(void* const* d_in, const int* in_sizes, int n_in,
                              void* d_out, int out_size, void* d_ws, size_t ws_size,
                              hipStream_t stream) {
    (void)in_sizes; (void)n_in; (void)out_size; (void)ws_size;

    const float* q_in = (const float*)d_in[0];
    const float* k_in = (const float*)d_in[1];
    const float* v_in = (const float*)d_in[2];
    const float* Wq = (const float*)d_in[3];  const float* bq = (const float*)d_in[4];
    const float* Wk = (const float*)d_in[5];  const float* bk = (const float*)d_in[6];
    const float* Wv = (const float*)d_in[7];  const float* bv = (const float*)d_in[8];
    const float* Wo = (const float*)d_in[9];  const float* bo = (const float*)d_in[10];

    const size_t TD = (size_t)TOK * D_MODEL;       // 6,291,456
    const size_t DD = (size_t)D_MODEL * D_MODEL;   //   589,824

    bf16* ws  = (bf16*)d_ws;
    bf16* xq  = ws;             // [T x D] bf16 activations
    bf16* xk  = xq  + TD;
    bf16* xv  = xk  + TD;
    bf16* wqb = xv  + TD;       // [D x D] bf16 weights
    bf16* wkb = wqb + DD;
    bf16* wvb = wkb + DD;
    bf16* wob = wvb + DD;
    bf16* Qb  = wob + DD;       // [B,H,S,Dk]
    bf16* Kb  = Qb  + TD;       // [B,H,S,Dk]
    bf16* Vb  = Kb  + TD;       // [B,H,Dk,S]  (pre-transposed)
    bf16* Ob  = Vb  + TD;       // [B,S,H*Dk]

    // 1) fp32 -> bf16 conversions
    {
        const int nTD = (int)TD, nDD = (int)DD;
        const int gTD = (nTD + 255) / 256, gDD = (nDD + 255) / 256;
        cvt_f32_bf16<<<gTD, 256, 0, stream>>>(q_in, xq, nTD);
        cvt_f32_bf16<<<gTD, 256, 0, stream>>>(k_in, xk, nTD);
        cvt_f32_bf16<<<gTD, 256, 0, stream>>>(v_in, xv, nTD);
        cvt_f32_bf16<<<gDD, 256, 0, stream>>>(Wq, wqb, nDD);
        cvt_f32_bf16<<<gDD, 256, 0, stream>>>(Wk, wkb, nDD);
        cvt_f32_bf16<<<gDD, 256, 0, stream>>>(Wv, wvb, nDD);
        cvt_f32_bf16<<<gDD, 256, 0, stream>>>(Wo, wob, nDD);
    }

    // 2) Q/K/V projections (Q pre-scaled by 1/sqrt(Dk) = 0.125; V transposed)
    dim3 ggrid(TOK / 128, D_MODEL / 64);
    gemm_bf16_kernel<<<ggrid, 256, 0, stream>>>(xq, wqb, bq, Qb,
                                                TOK, D_MODEL, D_MODEL, 0, 0.125f);
    gemm_bf16_kernel<<<ggrid, 256, 0, stream>>>(xk, wkb, bk, Kb,
                                                TOK, D_MODEL, D_MODEL, 0, 1.0f);
    gemm_bf16_kernel<<<ggrid, 256, 0, stream>>>(xv, wvb, bv, Vb,
                                                TOK, D_MODEL, D_MODEL, 2, 1.0f);

    // 3) flash attention (barrier-free)
    dim3 fgrid(SEQ / 128, NHEADS, BATCH);
    flash_attn_kernel<<<fgrid, 256, 0, stream>>>(Qb, Kb, Vb, Ob);

    // 4) output projection -> fp32 d_out
    gemm_bf16_kernel<<<ggrid, 256, 0, stream>>>(Ob, wob, bo, d_out,
                                                TOK, D_MODEL, D_MODEL, 1, 1.0f);
}
